// VSDN_IWAE_FILTER_32933809225804
// MI455X (gfx1250) — compile-verified, matrix operand order checked by reading, not run on hardware
//
#include <hip/hip_runtime.h>
#include <hip/hip_bf16.h>

// ---------------------------------------------------------------------------
// VSDN-IWAE fused latent-SDE scan for gfx1250 (MI455X, wave32, WMMA bf16)
//
// Pipeline (all on `stream`):
//   K0 conv_weights : fp32 weights -> bf16 WMMA-B-fragment-packed layout in ws
//   K1 z0_kernel    : z0 = tanh(MLP(cov)) broadcast over S
//   K2 scan_kernel  : 128 workgroups x 128 threads (4 waves). Each WG owns a
//                     16-row (b,s) tile and runs ALL 256 time steps with z and
//                     activations resident in LDS. 176 v_wmma per wave per step.
//   K3 loss_kernel  : IWAE + VAE reduction -> scalar loss
// ---------------------------------------------------------------------------

#define TT   256
#define BB   256
#define SS   8
#define BSR  2048          // B*S rows
#define DIN_ 64
#define DZ_  128
#define RR   128
#define HH   256

#define DT_     0.05f
#define SQDT_   0.22360679774997896f
#define LOG2PI_ 1.8378770664093453f

// packed bf16 weight-fragment bases (in uints) inside workspace
#define WD1_OFF 0u
#define WD2_OFF 32768u
#define WF1_OFF 49152u
#define WF2_OFF 65536u
#define WP1_OFF 81920u
#define WP2_OFF 114688u
#define WFRAG_UINTS 131072u   // 512 KB

typedef __attribute__((ext_vector_type(16))) __bf16 v16bf;
typedef __attribute__((ext_vector_type(8)))  float  v8f;

union Frag { unsigned u[8]; uint4 q[2]; v16bf v; };

__device__ __forceinline__ unsigned short bf16b(float x) {
  unsigned u = __float_as_uint(x);
  u += 0x7FFFu + ((u >> 16) & 1u);           // round-to-nearest-even
  return (unsigned short)(u >> 16);
}
__device__ __forceinline__ unsigned pkbf(float a, float b) {
  return (unsigned)bf16b(a) | ((unsigned)bf16b(b) << 16);
}
__device__ __forceinline__ v8f wmma_bf16(v16bf a, v16bf b, v8f c) {
  return __builtin_amdgcn_wmma_f32_16x16x32_bf16(false, a, false, b,
                                                 (short)0, c, false, false);
}

// ---------------------------------------------------------------------------
// K0: convert fp32 weights into bf16 fragments, packed per (ntile,ktile):
//   lane l holds column nn=l&15, k-values per the 16-bit B layout; 8 uints/lane
//   -> each wave fetches a 32x16 B tile with two b128 loads.
// ---------------------------------------------------------------------------
__global__ __launch_bounds__(256) void conv_weights(
    const float* __restrict__ Wd1, const float* __restrict__ Wd2,
    const float* __restrict__ Wf1, const float* __restrict__ Wf2,
    const float* __restrict__ Wp1, const float* __restrict__ Wp2,
    unsigned* __restrict__ wfrag) {
  unsigned idx = blockIdx.x * 256u + threadIdx.x;
  if (idx >= WFRAG_UINTS) return;
  const float* W; int K, N; unsigned base;
  if      (idx < WD2_OFF) { W = Wd1; K = 256; N = 256; base = WD1_OFF; }
  else if (idx < WF1_OFF) { W = Wd2; K = 256; N = 128; base = WD2_OFF; }
  else if (idx < WF2_OFF) { W = Wf1; K = 128; N = 256; base = WF1_OFF; }
  else if (idx < WP1_OFF) { W = Wf2; K = 256; N = 128; base = WF2_OFF; }
  else if (idx < WP2_OFF) { W = Wp1; K = 256; N = 256; base = WP1_OFF; }
  else                    { W = Wp2; K = 256; N = 128; base = WP2_OFF; }
  unsigned rel = idx - base;
  int Kt = K >> 5;
  int j    = rel & 7;
  int lane = (rel >> 3) & 31;
  int kt   = (int)((rel >> 8) % (unsigned)Kt);
  int nt   = (int)((rel >> 8) / (unsigned)Kt);
  int g = lane >> 4, nn = lane & 15;
  int kk = (j < 4) ? (g * 8 + 2 * j) : (16 + g * 8 + 2 * (j - 4));
  int krow = kt * 32 + kk;
  float a = W[(size_t)krow * N + nt * 16 + nn];
  float b = W[(size_t)(krow + 1) * N + nt * 16 + nn];
  wfrag[idx] = pkbf(a, b);
}

// ---------------------------------------------------------------------------
// K1: z0 = tanh(relu(cov@Wc1+bc1)@Wc2+bc2), broadcast over S samples
// ---------------------------------------------------------------------------
__global__ __launch_bounds__(128) void z0_kernel(
    const float* __restrict__ cov,
    const float* __restrict__ Wc1, const float* __restrict__ bc1,
    const float* __restrict__ Wc2, const float* __restrict__ bc2,
    float* __restrict__ z0) {
  __shared__ float sc[16];
  __shared__ float sh[64];
  int b = blockIdx.x, tid = threadIdx.x;
  if (tid < 16) sc[tid] = cov[b * 16 + tid];
  __syncthreads();
  if (tid < 64) {
    float a = bc1[tid];
    #pragma unroll
    for (int k = 0; k < 16; ++k) a += sc[k] * Wc1[k * 64 + tid];
    sh[tid] = a > 0.f ? a : 0.f;
  }
  __syncthreads();
  float a = bc2[tid];
  for (int k = 0; k < 64; ++k) a += sh[k] * Wc2[k * 128 + tid];
  a = tanhf(a);
  #pragma unroll
  for (int s = 0; s < SS; ++s)
    z0[((size_t)b * SS + s) * DZ_ + tid] = a;
}

// ---------------------------------------------------------------------------
// Fused 2-layer MLP for one 16-row tile, bf16 WMMA. L1: N=256 (4 N-tiles per
// wave), L2: N=128 (2 N-tiles per wave). MODE: 0 -> 0.1*tanh, 1 -> exp, 2 -> id
// ---------------------------------------------------------------------------
template <int KT1, int MODE>
__device__ __forceinline__ void mlp2(
    const unsigned* __restrict__ wf1, const float* __restrict__ b1,
    const unsigned* __restrict__ wf2, const float* __restrict__ b2,
    const unsigned (*sA)[128], unsigned short (*sHidH)[256],
    float (*outL)[128], int lane, int wave) {
  const int row = lane & 15, g = lane >> 4, m0 = g * 8;
  // ---- layer 1: [16,32*KT1] @ [32*KT1,256] -> relu -> bf16 hidden in LDS
  {
    const unsigned* Ar = sA[row];
    #pragma unroll
    for (int i = 0; i < 4; ++i) {
      const int nt = wave * 4 + i;
      v8f acc = {};
      #pragma unroll
      for (int kt = 0; kt < KT1; ++kt) {
        Frag a, bw;
        a.q[0] = *(const uint4*)&Ar[kt * 16 + g * 4];
        a.q[1] = *(const uint4*)&Ar[kt * 16 + 8 + g * 4];
        const uint4* wp = (const uint4*)&wf1[(((unsigned)nt * KT1 + kt) * 32 + lane) * 8u];
        bw.q[0] = wp[0]; bw.q[1] = wp[1];
        acc = wmma_bf16(a.v, bw.v, acc);
      }
      const float bb = b1[nt * 16 + row];
      #pragma unroll
      for (int v = 0; v < 8; ++v) {
        float x = acc[v] + bb;
        x = x > 0.f ? x : 0.f;
        sHidH[m0 + v][nt * 16 + row] = bf16b(x);
      }
    }
  }
  __syncthreads();
  // ---- layer 2: [16,256] @ [256,128] -> activation -> f32 in LDS
  {
    const unsigned* Hr = (const unsigned*)sHidH[row];
    #pragma unroll
    for (int i = 0; i < 2; ++i) {
      const int nt = wave * 2 + i;
      v8f acc = {};
      #pragma unroll
      for (int kt = 0; kt < 8; ++kt) {
        Frag a, bw;
        a.q[0] = *(const uint4*)&Hr[kt * 16 + g * 4];
        a.q[1] = *(const uint4*)&Hr[kt * 16 + 8 + g * 4];
        const uint4* wp = (const uint4*)&wf2[(((unsigned)nt * 8 + kt) * 32 + lane) * 8u];
        bw.q[0] = wp[0]; bw.q[1] = wp[1];
        acc = wmma_bf16(a.v, bw.v, acc);
      }
      const float bb = b2[nt * 16 + row];
      #pragma unroll
      for (int v = 0; v < 8; ++v) {
        float x = acc[v] + bb;
        if (MODE == 0)      x = 0.1f * tanhf(x);
        else if (MODE == 1) x = __expf(x);
        outL[m0 + v][nt * 16 + row] = x;
      }
    }
  }
  __syncthreads();
}

// ---------------------------------------------------------------------------
// K2: persistent scan. One WG per 16 (b,s) rows, all 256 steps in LDS.
// ---------------------------------------------------------------------------
__global__ __launch_bounds__(128) void scan_kernel(
    const float* __restrict__ X,  const float* __restrict__ M,
    const float* __restrict__ noise,
    const float* __restrict__ path_h, const float* __restrict__ path_hpos,
    const float* __restrict__ bd1, const float* __restrict__ bd2,
    const float* __restrict__ bf1, const float* __restrict__ bf2,
    const float* __restrict__ bp1, const float* __restrict__ bp2,
    const unsigned* __restrict__ wfrag, const float* __restrict__ z0,
    float* __restrict__ z_out, float* __restrict__ neg_lll,
    float* __restrict__ neg_lwk,
    float* __restrict__ lwk_ws, float* __restrict__ lll_ws,
    float* __restrict__ kld_ws) {
  __shared__ unsigned       sA[16][128];     // [h|z] bf16-pair A operand (K=256)
  __shared__ unsigned short sHidH[16][256];  // L1 activations (bf16)
  __shared__ float sPrior[16][128];
  __shared__ float sPoster[16][128];
  __shared__ float sDiff[16][128];
  __shared__ float sP[16][128];
  __shared__ float sZ[16][128];
  __shared__ float sPmax[16][8], sPd[16][8], sPw[16][8];
  __shared__ float sZmax[16];
  __shared__ float sAcc[16][3];              // log_wk, log_ll, kld

  const int tid = threadIdx.x, lane = tid & 31, wave = tid >> 5;
  const int r0 = blockIdx.x * 16;

  for (int i = tid; i < 16 * 128; i += 128)
    sZ[i >> 7][i & 127] = z0[(size_t)(r0 + (i >> 7)) * 128 + (i & 127)];
  if (tid < 16) { sAcc[tid][0] = 0.f; sAcc[tid][1] = 0.f; sAcc[tid][2] = 0.f; }
  __syncthreads();

  const unsigned* wd1 = wfrag + WD1_OFF; const unsigned* wd2 = wfrag + WD2_OFF;
  const unsigned* wf1w = wfrag + WF1_OFF; const unsigned* wf2w = wfrag + WF2_OFF;
  const unsigned* wp1 = wfrag + WP1_OFF; const unsigned* wp2 = wfrag + WP2_OFF;

  for (int t = 0; t < TT; ++t) {
    // ---- pack A z-half (K 128..255) from current z, h-half from path_hpos
    for (int i = tid; i < 1024; i += 128) {
      int m = i >> 6, q = i & 63;
      sA[m][64 + q] = pkbf(sZ[m][2 * q], sZ[m][2 * q + 1]);
      int b = (r0 + m) >> 3;
      const float* hr = path_hpos + ((size_t)t * BB + b) * RR;
      sA[m][q] = pkbf(hr[2 * q], hr[2 * q + 1]);
    }
    __syncthreads();
    mlp2<8, 0>(wd1, bd1, wd2, bd2, sA, sHidH, sPoster, lane, wave);

    // ---- swap h-half to path_h (z-half unchanged)
    for (int i = tid; i < 1024; i += 128) {
      int m = i >> 6, q = i & 63;
      int b = (r0 + m) >> 3;
      const float* hr = path_h + ((size_t)t * BB + b) * RR;
      sA[m][q] = pkbf(hr[2 * q], hr[2 * q + 1]);
      if (t + 1 < TT)  // warm L2 for next step's h
        __builtin_prefetch(path_h + ((size_t)(t + 1) * BB + b) * RR + 2 * q, 0, 1);
    }
    __syncthreads();
    mlp2<8, 0>(wd1, bd1, wd2, bd2, sA, sHidH, sPrior, lane, wave);   // prior
    mlp2<4, 1>(wf1w, bf1, wf2w, bf2, sA, sHidH, sDiff, lane, wave);  // exp diffusion

    // ---- Euler–Maruyama z update + err accumulators (row m = tid>>3)
    {
      const int m = tid >> 3, c = tid & 7;
      const float* nz = noise + ((size_t)t * BSR + (r0 + m)) * DZ_;
      float mx = -3.4e38f, dts = 0.f, dws = 0.f;
      #pragma unroll
      for (int j = 0; j < 16; ++j) {
        int d = c * 16 + j;
        float df = sDiff[m][d], po = sPoster[m][d], pr = sPrior[m][d];
        float nv = nz[d];
        float zn = sZ[m][d] + DT_ * po + SQDT_ * df * nv;
        sZ[m][d] = zn;
        mx = fmaxf(mx, zn);
        float err = (pr - po) / df;
        dts += 0.5f * DT_ * err * err;
        dws += SQDT_ * err * nv;
      }
      sPmax[m][c] = mx; sPd[m][c] = dts; sPw[m][c] = dws;
    }
    __syncthreads();
    if (tid < 16) {
      float mx = sPmax[tid][0], dt = sPd[tid][0], dw = sPw[tid][0];
      #pragma unroll
      for (int j = 1; j < 8; ++j) {
        mx = fmaxf(mx, sPmax[tid][j]); dt += sPd[tid][j]; dw += sPw[tid][j];
      }
      sZmax[tid] = mx;
      sAcc[tid][0] -= (dt + dw);   // log_wk
      sAcc[tid][2] += dt;          // kld
    }
    __syncthreads();

    // ---- pack A z-half = z/zmax (h-half still path_h) -> p model
    for (int i = tid; i < 1024; i += 128) {
      int m = i >> 6, q = i & 63;
      float zm = sZmax[m];
      sA[m][64 + q] = pkbf(sZ[m][2 * q] / zm, sZ[m][2 * q + 1] / zm);
    }
    __syncthreads();
    mlp2<8, 2>(wp1, bp1, wp2, bp2, sA, sHidH, sP, lane, wave);

    // ---- masked Gaussian NLL over DIN=64 dims
    {
      const int m = tid >> 3, c = tid & 7;
      const int b = (r0 + m) >> 3;
      const float* Xt = X + ((size_t)t * BB + b) * DIN_;
      const float* Mt = M + ((size_t)t * BB + b) * DIN_;
      float nl = 0.f;
      #pragma unroll
      for (int j = 0; j < 8; ++j) {
        int d = c * 8 + j;
        float mean = sP[m][d], logv = sP[m][64 + d];
        float e = Xt[d] - mean;
        nl += 0.5f * (LOG2PI_ + logv + e * e * __expf(-logv)) * Mt[d];
      }
      sPd[m][c] = nl;
    }
    __syncthreads();
    if (tid < 16) {
      float nl = sPd[tid][0];
      #pragma unroll
      for (int j = 1; j < 8; ++j) nl += sPd[tid][j];
      sAcc[tid][1] -= nl;          // log_ll
    }
    __syncthreads();
  }

  // ---- write results
  for (int i = tid; i < 16 * 128; i += 128)
    z_out[(size_t)(r0 + (i >> 7)) * 128 + (i & 127)] = sZ[i >> 7][i & 127];
  if (tid < 16) {
    int r = r0 + tid;
    lwk_ws[r] = sAcc[tid][0];
    lll_ws[r] = sAcc[tid][1];
    kld_ws[r] = sAcc[tid][2];
    neg_lll[r] = -sAcc[tid][1];
    neg_lwk[r] = -sAcc[tid][0];
  }
}

// ---------------------------------------------------------------------------
// K3: IWAE + VAE loss reduction
// ---------------------------------------------------------------------------
__global__ __launch_bounds__(256) void loss_kernel(
    const float* __restrict__ lwk, const float* __restrict__ lll,
    const float* __restrict__ kld, float* __restrict__ out_loss) {
  __shared__ float sI[256], sV[256];
  int b = threadIdx.x;
  float lw[SS], mx = -3.4e38f, vsum = 0.f;
  #pragma unroll
  for (int s = 0; s < SS; ++s) {
    lw[s] = lwk[b * SS + s] + lll[b * SS + s];
    mx = fmaxf(mx, lw[s]);
    vsum += kld[b * SS + s] - lll[b * SS + s];   // LAMBDA = 1
  }
  float sw = 0.f, ws_[SS];
  #pragma unroll
  for (int s = 0; s < SS; ++s) { ws_[s] = __expf(lw[s] - mx); sw += ws_[s]; }
  float iw = 0.f;
  #pragma unroll
  for (int s = 0; s < SS; ++s) iw += (ws_[s] / sw) * lw[s];
  sI[b] = -iw;
  sV[b] = vsum;
  __syncthreads();
  if (b == 0) {
    float a = 0.f, c = 0.f;
    for (int i = 0; i < 256; ++i) { a += sI[i]; c += sV[i]; }
    float loss_iwae = a / 256.0f;
    float loss_vae  = c / (256.0f * (float)SS);
    out_loss[0] = 0.5f * loss_iwae + 0.5f * loss_vae;  // ALPHA = 0.5
  }
}

// ---------------------------------------------------------------------------
extern "C" void kernel_launch(void* const* d_in, const int* in_sizes, int n_in,
                              void* d_out, int out_size, void* d_ws, size_t ws_size,
                              hipStream_t stream) {
  (void)in_sizes; (void)n_in; (void)out_size; (void)ws_size;
  const float* X     = (const float*)d_in[0];
  const float* M     = (const float*)d_in[1];
  const float* cov   = (const float*)d_in[2];
  const float* noise = (const float*)d_in[3];
  const float* ph    = (const float*)d_in[4];
  const float* php   = (const float*)d_in[5];
  const float* Wd1 = (const float*)d_in[6];  const float* bd1 = (const float*)d_in[7];
  const float* Wd2 = (const float*)d_in[8];  const float* bd2 = (const float*)d_in[9];
  const float* Wf1 = (const float*)d_in[10]; const float* bf1 = (const float*)d_in[11];
  const float* Wf2 = (const float*)d_in[12]; const float* bf2 = (const float*)d_in[13];
  const float* Wp1 = (const float*)d_in[14]; const float* bp1 = (const float*)d_in[15];
  const float* Wp2 = (const float*)d_in[16]; const float* bp2 = (const float*)d_in[17];
  const float* Wc1 = (const float*)d_in[18]; const float* bc1 = (const float*)d_in[19];
  const float* Wc2 = (const float*)d_in[20]; const float* bc2 = (const float*)d_in[21];

  unsigned char* ws = (unsigned char*)d_ws;
  unsigned* wfrag = (unsigned*)ws;                               // 512 KB
  float* z0   = (float*)(ws + 512 * 1024);                        // 1 MB
  float* lwk  = (float*)(ws + 512 * 1024 + 1024 * 1024);          // 8 KB
  float* lll  = lwk + BSR;
  float* kld  = lll + BSR;

  float* out    = (float*)d_out;
  float* z_out  = out;                 // [B,S,DZ] = 262144
  float* loss   = out + 262144;        // scalar
  float* nlll   = out + 262145;        // [B,S]
  float* nlwk   = nlll + BSR;          // [B,S]

  hipLaunchKernelGGL(conv_weights, dim3(512), dim3(256), 0, stream,
                     Wd1, Wd2, Wf1, Wf2, Wp1, Wp2, wfrag);
  hipLaunchKernelGGL(z0_kernel, dim3(256), dim3(128), 0, stream,
                     cov, Wc1, bc1, Wc2, bc2, z0);
  hipLaunchKernelGGL(scan_kernel, dim3(128), dim3(128), 0, stream,
                     X, M, noise, ph, php, bd1, bd2, bf1, bf2, bp1, bp2,
                     wfrag, z0, z_out, nlll, nlwk, lwk, lll, kld);
  hipLaunchKernelGGL(loss_kernel, dim3(1), dim3(256), 0, stream,
                     lwk, lll, kld, loss);
}
